// Att_AdaIn_18949395710001
// MI455X (gfx1250) — matrix operand order repaired and checked by reading, not compiled, not run on hardware
//
#include <hip/hip_runtime.h>

// ---------------------------------------------------------------------------
// Att_AdaIn for MI455X (gfx1250): bf16 WMMA flash-attention pipeline with
// CDNA5 async global->LDS DMA and double-buffered K/V tiles.
//   B=4, C=256, H=W=64, N=4096.  ~77 GFLOP, compute-bound.
// ---------------------------------------------------------------------------

#define NB    4
#define CCH   256
#define NPIX  4096

typedef __bf16 bf16_t;
typedef bf16_t v16bf __attribute__((ext_vector_type(16)));
typedef float  v8f   __attribute__((ext_vector_type(8)));

__device__ __forceinline__ unsigned short f2bf(float f) {
  unsigned int u = __float_as_uint(f);
  u += 0x7FFFu + ((u >> 16) & 1u);           // round-to-nearest-even
  return (unsigned short)(u >> 16);
}
__device__ __forceinline__ float bf2f(unsigned short h) {
  return __uint_as_float(((unsigned int)h) << 16);
}

// Low 32 bits of a generic pointer to __shared__ == wave-relative LDS offset
// (AMDGPU local->flat addrspacecast is {aperture_hi, zext(lds_offset)}).
__device__ __forceinline__ unsigned lds_off(const void* p) {
  return (unsigned)(unsigned long long)p;
}

// CDNA5 async DMA: global -> LDS, 16 bytes per lane, tracked by ASYNCcnt.
__device__ __forceinline__ void async_copy_b128(unsigned lds_byte_off, const void* g) {
  asm volatile("global_load_async_to_lds_b128 %0, %1, off"
               :: "v"(lds_byte_off), "v"(g) : "memory");
}
__device__ __forceinline__ void wait_async0() {
  asm volatile("s_wait_asynccnt 0x0" ::: "memory");
}

// A-fragment 16x32 bf16 (ISA 7.12.2): lanes 0-15 rows 0-15 hold K pairs
// {0,1},{2,3},{4,5},{6,7},{16,17},{18,19},{20,21},{22,23}; lanes 16-31 same
// rows with K offset +8 / +24.  base -> (row0, k0) in LDS, pitch in halves.
__device__ __forceinline__ v16bf lds_fragA(const unsigned short* base, int pitch) {
  const int lane = threadIdx.x & 31;
  const unsigned short* p = base + (lane & 15) * pitch + ((lane & 16) ? 8 : 0);
  v16bf a;
#pragma unroll
  for (int v = 0; v < 8; ++v) {
    const int k = (v < 4) ? (2 * v) : (8 + 2 * v);          // 0,2,4,6,16,18,20,22
    const unsigned int u = *reinterpret_cast<const unsigned int*>(p + k);
    a[2 * v]     = __builtin_bit_cast(bf16_t, (unsigned short)(u & 0xFFFFu));
    a[2 * v + 1] = __builtin_bit_cast(bf16_t, (unsigned short)(u >> 16));
  }
  return a;
}

// B-fragment 32x16 bf16: lane holds column (lane&15), same K order as A-fragment
// (lanes 0-15 K 0-7,16-23; lanes 16-31 K 8-15,24-31).  LDS stores B transposed as
// [col][k] so each lane reads two runs of 8 halves along k.  base -> (col0, k0).
__device__ __forceinline__ v16bf lds_fragB(const unsigned short* base, int pitch) {
  const int lane = threadIdx.x & 31;
  const unsigned short* p = base + (lane & 15) * pitch + ((lane & 16) ? 8 : 0);
  v16bf b;
#pragma unroll
  for (int e = 0; e < 8; ++e) {
    const unsigned int u = *reinterpret_cast<const unsigned int*>(p + ((e < 4) ? (2 * e) : (8 + 2 * e)));
    b[2 * e]     = __builtin_bit_cast(bf16_t, (unsigned short)(u & 0xFFFFu));
    b[2 * e + 1] = __builtin_bit_cast(bf16_t, (unsigned short)(u >> 16));
  }
  return b;
}

#define WMMA_BF16(A, B, C) \
  __builtin_amdgcn_wmma_f32_16x16x32_bf16(false, (A), false, (B), (short)0, (C), false, false)

// C/D frag (16x16 f32): row = v + ((lane&16)?8:0), col = lane&15.

// ---------------------------------------------------------------------------
// Kernel 1: q/k/v projections.  grid = (N/64, B, 3); 256 threads (8 waves).
//   Q,K -> [b][n][c] bf16 ; V -> [b][c][n] bf16 (attention-friendly layout).
// ---------------------------------------------------------------------------
#define XP 264   // pitch (halves) of 64x256 activation tile
#define WP 40    // pitch (halves) of 256x32 weight slab

__global__ __launch_bounds__(256) void qkv_proj_kernel(
    const float* __restrict__ x, const float* __restrict__ y,
    const float* __restrict__ Wq, const float* __restrict__ bq,
    const float* __restrict__ Wk, const float* __restrict__ bk,
    const float* __restrict__ Wv, const float* __restrict__ bv,
    unsigned short* __restrict__ Qh, unsigned short* __restrict__ Kh,
    unsigned short* __restrict__ Vh)
{
  __shared__ unsigned short Xs[64 * XP];     // X^T tile [pixel][channel]
  __shared__ unsigned short Wch[256 * WP];   // W^T slab [o][c-chunk]

  const int tid   = threadIdx.x;
  const int b     = blockIdx.y;
  const int which = blockIdx.z;              // 0:q(x) 1:k(y) 2:v(y)
  const int n0    = blockIdx.x * 64;

  const float* src  = (which == 0) ? x  : y;
  const float* W    = (which == 0) ? Wq : (which == 1) ? Wk : Wv;
  const float* bias = (which == 0) ? bq : (which == 1) ? bk : bv;

  // stage X^T tile (coalesced along n), f32 -> bf16
  for (int idx = tid; idx < 64 * 256; idx += 256) {
    const int c = idx >> 6, p = idx & 63;
    Xs[p * XP + c] = f2bf(src[(((b << 8) + c) << 12) + n0 + p]);
  }
  __syncthreads();

  const int lane = tid & 31, w = tid >> 5;
  const int r0 = (w & 3) * 16;        // pixel rows of this wave
  const int o0 = (w >> 2) * 128;      // output-channel cols of this wave

  v8f acc[8];
#pragma unroll
  for (int f = 0; f < 8; ++f)
#pragma unroll
    for (int v = 0; v < 8; ++v) acc[f][v] = 0.0f;

  for (int kc = 0; kc < 8; ++kc) {
    for (int idx = tid; idx < 256 * 32; idx += 256) {
      const int o = idx >> 5, cc = idx & 31;
      Wch[o * WP + cc] = f2bf(W[(o << 8) + (kc << 5) + cc]);
    }
    __syncthreads();
    const v16bf A = lds_fragA(&Xs[r0 * XP + (kc << 5)], XP);
#pragma unroll
    for (int f = 0; f < 8; ++f) {
      const v16bf B = lds_fragB(&Wch[(o0 + 16 * f) * WP], WP);
      acc[f] = WMMA_BF16(A, B, acc[f]);
    }
    __syncthreads();    // protect Wch before next slab load
  }

  // bias epilogue (bias depends on column only)
#pragma unroll
  for (int f = 0; f < 8; ++f) {
    const float bv_ = bias[o0 + 16 * f + (lane & 15)];
#pragma unroll
    for (int v = 0; v < 8; ++v) acc[f][v] += bv_;
  }

  if (which < 2) {                       // Q/K: [b][n][c], 32B runs per row
    unsigned short* dst = (which == 0) ? Qh : Kh;
#pragma unroll
    for (int f = 0; f < 8; ++f) {
      const int o = o0 + 16 * f + (lane & 15);
#pragma unroll
      for (int v = 0; v < 8; ++v) {
        const int n = n0 + r0 + v + ((lane & 16) ? 8 : 0);
        dst[(((b << 12) + n) << 8) + o] = f2bf(acc[f][v]);
      }
    }
  } else {                               // V: [b][c][n], restage via LDS
    __syncthreads();
#pragma unroll
    for (int f = 0; f < 8; ++f) {
      const int o = o0 + 16 * f + (lane & 15);
#pragma unroll
      for (int v = 0; v < 8; ++v) {
        const int p = r0 + v + ((lane & 16) ? 8 : 0);
        Xs[o * 66 + p] = f2bf(acc[f][v]);   // 256*66 == 64*XP, exact reuse
      }
    }
    __syncthreads();
    for (int idx = tid; idx < 64 * 256; idx += 256) {
      const int o = idx >> 6, p = idx & 63;
      Vh[(((b << 8) + o) << 12) + n0 + p] = Xs[o * 66 + p];
    }
  }
}

// ---------------------------------------------------------------------------
// Kernel 2: flash attention.  grid = (N/64, B); 256 threads (8 waves).
//   Double-buffered K/V tiles fetched with global_load_async_to_lds_b128
//   (ASYNCcnt) so tile t+1 DMA overlaps tile t WMMA work.
// ---------------------------------------------------------------------------
#define QP 264   // Q/K tile pitch (halves); row stride 528B (16B-aligned)
#define VP 72    // V^T tile pitch (halves); row stride 144B (16B-aligned)
#define SP 66    // score tile pitch (floats)
#define PP 72    // P tile pitch (halves)

__global__ __launch_bounds__(256) void attention_kernel(
    const unsigned short* __restrict__ Qh, const unsigned short* __restrict__ Kh,
    const unsigned short* __restrict__ Vh, unsigned short* __restrict__ Zh)
{
  __shared__ unsigned short Qs[64 * QP];
  __shared__ unsigned short Ks[2][64 * QP];    // double-buffered key tiles
  __shared__ unsigned short Vts[2][256 * VP];  // double-buffered V^T tiles
  __shared__ float          Ss[64 * SP];
  __shared__ unsigned short Ps[64 * PP];
  __shared__ float mrow[64], lrow[64], arow[64];
  __shared__ float redM[256], redS[256];

  const int tid  = threadIdx.x;
  const int b    = blockIdx.y;
  const int q0   = blockIdx.x * 64;
  const int lane = tid & 31, w = tid >> 5;

  // issue async DMA for one K/V tile into buffer `buf` (16 b128 per thread)
  auto issue_tile = [&](int kt, int buf) {
    const int k0 = kt << 6;
    for (int idx = tid; idx < 2048; idx += 256) {            // K: 64r x 512B
      const int kk = idx >> 5, c16 = idx & 31;
      async_copy_b128(lds_off(&Ks[buf][kk * QP + c16 * 8]),
                      &Kh[(((b << 12) + k0 + kk) << 8) + c16 * 8]);
    }
    for (int idx = tid; idx < 2048; idx += 256) {            // V^T: 256c x 128B
      const int c = idx >> 3, ch = idx & 7;
      async_copy_b128(lds_off(&Vts[buf][c * VP + ch * 8]),
                      &Vh[(((b << 8) + c) << 12) + k0 + ch * 8]);
    }
  };

  // stage Q tile pre-scaled by 1/sqrt(C) = 1/16 (needs conversion -> sync path)
  for (int idx = tid; idx < 64 * 256; idx += 256) {
    const int c = idx & 255, r = idx >> 8;
    Qs[r * QP + c] = f2bf(bf2f(Qh[(((b << 12) + q0 + r) << 8) + c]) * 0.0625f);
  }
  if (tid < 64) { mrow[tid] = -3.0e38f; lrow[tid] = 0.0f; }

  const int orow = (w & 3) * 16;     // query rows of this wave's O tile
  const int ocol = (w >> 2) * 128;   // channel cols of this wave's O tile
  v8f acc[8];
#pragma unroll
  for (int f = 0; f < 8; ++f)
#pragma unroll
    for (int v = 0; v < 8; ++v) acc[f][v] = 0.0f;

  issue_tile(0, 0);                  // preload first tile

  for (int kt = 0; kt < 64; ++kt) {
    const int cur = kt & 1;
    wait_async0();                   // this wave's DMA into buf `cur` landed
    __syncthreads();                 // all waves' DMA visible; prev iter done
    if (kt + 1 < 64) issue_tile(kt + 1, cur ^ 1);   // overlap next tile DMA

    // ---- S = Q K^T : each wave computes a 16x32 slice (2 frags) ----
    const int sr = (w & 3) * 16;
    const int sc = (w >> 2) * 32;
    v8f s0, s1;
#pragma unroll
    for (int v = 0; v < 8; ++v) { s0[v] = 0.0f; s1[v] = 0.0f; }
#pragma unroll
    for (int kc = 0; kc < 8; ++kc) {
      const v16bf A  = lds_fragA(&Qs[sr * QP + (kc << 5)], QP);
      const v16bf B0 = lds_fragB(&Ks[cur][sc * QP + (kc << 5)], QP);
      const v16bf B1 = lds_fragB(&Ks[cur][(sc + 16) * QP + (kc << 5)], QP);
      s0 = WMMA_BF16(A, B0, s0);
      s1 = WMMA_BF16(A, B1, s1);
    }
    {
      const int col = sc + (lane & 15);
#pragma unroll
      for (int v = 0; v < 8; ++v) {
        const int row = sr + v + ((lane & 16) ? 8 : 0);
        Ss[row * SP + col]      = s0[v];
        Ss[row * SP + col + 16] = s1[v];
      }
    }
    __syncthreads();

    // ---- online softmax: 4 threads per query row ----
    const int r = tid & 63, qd = tid >> 6;
    {
      float mx = -3.0e38f;
#pragma unroll
      for (int k = 0; k < 16; ++k) mx = fmaxf(mx, Ss[r * SP + qd * 16 + k]);
      redM[qd * 64 + r] = mx;
    }
    __syncthreads();
    const float mo = mrow[r];
    const float mn = fmaxf(fmaxf(fmaxf(redM[r], redM[64 + r]),
                                 fmaxf(redM[128 + r], redM[192 + r])), mo);
    {
      float sum = 0.0f;
#pragma unroll
      for (int k = 0; k < 16; ++k) {
        const float p = __expf(Ss[r * SP + qd * 16 + k] - mn);
        Ps[r * PP + qd * 16 + k] = f2bf(p);
        sum += p;
      }
      redS[qd * 64 + r] = sum;
    }
    __syncthreads();
    if (qd == 0) {
      const float s4 = redS[r] + redS[64 + r] + redS[128 + r] + redS[192 + r];
      const float al = __expf(mo - mn);
      lrow[r] = lrow[r] * al + s4;
      arow[r] = al;
      mrow[r] = mn;
    }
    __syncthreads();

    // ---- O = O*alpha + P · V ----
    float a8[8];
#pragma unroll
    for (int v = 0; v < 8; ++v) a8[v] = arow[orow + v + ((lane & 16) ? 8 : 0)];
#pragma unroll
    for (int f = 0; f < 8; ++f)
#pragma unroll
      for (int v = 0; v < 8; ++v) acc[f][v] *= a8[v];
#pragma unroll
    for (int ks = 0; ks < 2; ++ks) {
      const v16bf A = lds_fragA(&Ps[orow * PP + (ks << 5)], PP);
#pragma unroll
      for (int f = 0; f < 8; ++f) {
        const v16bf B = lds_fragB(&Vts[cur][(ocol + 16 * f) * VP + (ks << 5)], VP);
        acc[f] = WMMA_BF16(A, B, acc[f]);
      }
    }
    // top-of-loop barrier protects Ks/Vts/Ps/Ss before reuse
  }

  // finalize: O /= l, store Z' [b][n][c] bf16
  float li[8];
#pragma unroll
  for (int v = 0; v < 8; ++v) li[v] = 1.0f / lrow[orow + v + ((lane & 16) ? 8 : 0)];
#pragma unroll
  for (int f = 0; f < 8; ++f) {
    const int c = ocol + 16 * f + (lane & 15);
#pragma unroll
    for (int v = 0; v < 8; ++v) {
      const int n = q0 + orow + v + ((lane & 16) ? 8 : 0);
      Zh[(((b << 12) + n) << 8) + c] = f2bf(acc[f][v] * li[v]);
    }
  }
}

// ---------------------------------------------------------------------------
// Kernel 3: out = x + Wo·z + bo.  grid = (N/64, B); 256 threads.
//   Z tile staged with async DMA (plain bf16 copy).
// ---------------------------------------------------------------------------
__global__ __launch_bounds__(256) void out_proj_kernel(
    const unsigned short* __restrict__ Zh, const float* __restrict__ Wo,
    const float* __restrict__ bo, const float* __restrict__ x,
    float* __restrict__ out)
{
  __shared__ unsigned short Zs[64 * XP];
  __shared__ unsigned short Wch[256 * WP];
  __shared__ float Os[256 * 66];             // f32 epilogue staging [o][p]

  const int tid = threadIdx.x;
  const int b   = blockIdx.y;
  const int n0  = blockIdx.x * 64;

  for (int idx = tid; idx < 2048; idx += 256) {   // async-stage Z tile
    const int r = idx >> 5, c16 = idx & 31;
    async_copy_b128(lds_off(&Zs[r * XP + c16 * 8]),
                    &Zh[(((b << 12) + n0 + r) << 8) + c16 * 8]);
  }
  wait_async0();
  __syncthreads();

  const int lane = tid & 31, w = tid >> 5;
  const int r0 = (w & 3) * 16;
  const int o0 = (w >> 2) * 128;

  v8f acc[8];
#pragma unroll
  for (int f = 0; f < 8; ++f)
#pragma unroll
    for (int v = 0; v < 8; ++v) acc[f][v] = 0.0f;

  for (int kc = 0; kc < 8; ++kc) {
    for (int idx = tid; idx < 256 * 32; idx += 256) {
      const int o = idx >> 5, cc = idx & 31;
      Wch[o * WP + cc] = f2bf(Wo[(o << 8) + (kc << 5) + cc]);
    }
    __syncthreads();
    const v16bf A = lds_fragA(&Zs[r0 * XP + (kc << 5)], XP);
#pragma unroll
    for (int f = 0; f < 8; ++f) {
      const v16bf B = lds_fragB(&Wch[(o0 + 16 * f) * WP], WP);
      acc[f] = WMMA_BF16(A, B, acc[f]);
    }
    __syncthreads();
  }

  // bias + stage f32 result [o][p] for coalesced residual store
#pragma unroll
  for (int f = 0; f < 8; ++f) {
    const int o = o0 + 16 * f + (lane & 15);
    const float bv_ = bo[o];
#pragma unroll
    for (int v = 0; v < 8; ++v) {
      const int p = r0 + v + ((lane & 16) ? 8 : 0);
      Os[o * 66 + p] = acc[f][v] + bv_;
    }
  }
  __syncthreads();
  for (int idx = tid; idx < 64 * 256; idx += 256) {
    const int o = idx >> 6, p = idx & 63;
    const int g = (((b << 8) + o) << 12) + n0 + p;
    out[g] = x[g] + Os[o * 66 + p];
  }
}

// ---------------------------------------------------------------------------
extern "C" void kernel_launch(void* const* d_in, const int* in_sizes, int n_in,
                              void* d_out, int out_size, void* d_ws, size_t ws_size,
                              hipStream_t stream) {
  const float* x  = (const float*)d_in[0];
  const float* y  = (const float*)d_in[1];
  const float* Wq = (const float*)d_in[2];
  const float* bq = (const float*)d_in[3];
  const float* Wk = (const float*)d_in[4];
  const float* bk = (const float*)d_in[5];
  const float* Wv = (const float*)d_in[6];
  const float* bv = (const float*)d_in[7];
  const float* Wo = (const float*)d_in[8];
  const float* bo = (const float*)d_in[9];
  float* out = (float*)d_out;

  const size_t perTensor = (size_t)NB * NPIX * CCH;          // bf16 elements
  unsigned short* Qh = (unsigned short*)d_ws;
  unsigned short* Kh = Qh + perTensor;
  unsigned short* Vh = Kh + perTensor;
  unsigned short* Zh = Vh + perTensor;                        // 32 MB total

  qkv_proj_kernel<<<dim3(NPIX / 64, NB, 3), 256, 0, stream>>>(
      x, y, Wq, bq, Wk, bk, Wv, bv, Qh, Kh, Vh);
  attention_kernel<<<dim3(NPIX / 64, NB), 256, 0, stream>>>(Qh, Kh, Vh, Zh);
  out_proj_kernel<<<dim3(NPIX / 64, NB), 256, 0, stream>>>(Zh, Wo, bo, x, out);
}
